// Clustering_20435454394868
// MI455X (gfx1250) — matrix operand order repaired and checked by reading, not compiled, not run on hardware
//
#include <hip/hip_runtime.h>
#include <math.h>

// ---------------- problem constants ----------------
#define B_    4
#define H_    16
#define L_    4096
#define DK_   64
#define LK0_  4096
#define M_    8          // LOG_LK == LOG_L
#define NC_   3
#define BH_   (B_ * H_)          // 64
#define SVROW_ (M_ * DK_)        // 512 floats per (b,h) of a shrunken tensor
#define SPLIT_ 8                 // k-split for the shrink reduction

typedef __attribute__((ext_vector_type(4)))  float    v4f;
typedef __attribute__((ext_vector_type(8)))  float    v8f;
typedef __attribute__((ext_vector_type(16))) _Float16 v16h;

// ---------------- workspace layout (float slots) ----------------
// partials: [3 tensors][BH][SPLIT][M][DK]
#define PART_N   (3 * BH_ * SPLIT_ * SVROW_)     // 786432 floats
#define SV_OFF   (PART_N)                        // shrunken Ks/Vs/Qs: [3][BH][M][DK]
#define SV_N     (3 * BH_ * SVROW_)              // 98304 floats
#define SCAT_OFF (SV_OFF + SV_N)                 // scat: [NC][H][M]
#define SCAT_N   (NC_ * H_ * M_)                 // 384 floats
#define INDS_OFF (SCAT_OFF + SCAT_N)             // int inds_q[4], inds_k[4]

__device__ __forceinline__ void wait_asynccnt0() {
#if __has_builtin(__builtin_amdgcn_s_wait_asynccnt)
  __builtin_amdgcn_s_wait_asynccnt(0);
#else
  asm volatile("s_wait_asynccnt 0x0" ::: "memory");
#endif
}

// =====================================================================
// Kernel 1: streaming shrink partial-reduction.
//   out[bh, split, m, d] = sum_{k in split-range} W[m,k] * X[bh, k, d]
// Memory-bound: b128 loads of X; weights staged into LDS with CDNA5
// GLOBAL_LOAD_ASYNC_TO_LDS_B32 (ASYNCcnt), freeing VALU/VGPR slots for
// the 192 MiB X stream.
// =====================================================================
__global__ void k_shrink_partial(const float* __restrict__ X,
                                 const float* __restrict__ W,
                                 float* __restrict__ part) {
  __shared__ float wsm[M_ * 256];          // 8 KB weight stage
  __shared__ float red[16 * M_ * DK_];     // 32 KB cross-group reduce

  const int bh = blockIdx.x;
  const int sp = blockIdx.y;
  const int tid = threadIdx.x;
  const int d4 = tid & 15;                 // float4 lane over d
  const int kq = tid >> 4;                 // 16 k-groups

  v4f acc[M_];
#pragma unroll
  for (int m = 0; m < M_; ++m) { acc[m][0] = 0.f; acc[m][1] = 0.f; acc[m][2] = 0.f; acc[m][3] = 0.f; }

  const float* xb = X + (size_t)bh * LK0_ * DK_;
  const unsigned lds_base = (unsigned)(unsigned long long)(uintptr_t)&wsm[tid];

  for (int ch = 0; ch < 2; ++ch) {
    const int k0 = sp * 512 + ch * 256;
    // async-stage the 256-wide weight chunk for all 8 output rows:
    // each lane copies one dword global->LDS, no VGPR round-trip.
#pragma unroll
    for (int m = 0; m < M_; ++m) {
      const unsigned ldsa = lds_base + (unsigned)(m * 256 * 4);
      const unsigned long long ga =
          (unsigned long long)(uintptr_t)(W + (size_t)m * LK0_ + k0 + tid);
      asm volatile("global_load_async_to_lds_b32 %0, %1, off"
                   :: "v"(ldsa), "v"(ga) : "memory");
    }
    wait_asynccnt0();          // s_wait_asynccnt 0
    __syncthreads();

    for (int kk = kq; kk < 256; kk += 16) {
      const float* xp = xb + (size_t)(k0 + kk) * DK_ + d4 * 4;
      __builtin_prefetch(xp + 16 * DK_, 0, 1);          // global_prefetch_b8
      v4f x = *(const v4f*)xp;                          // global_load_b128
#pragma unroll
      for (int m = 0; m < M_; ++m) {
        const float w = wsm[m * 256 + kk];
        acc[m][0] += w * x[0]; acc[m][1] += w * x[1];
        acc[m][2] += w * x[2]; acc[m][3] += w * x[3];
      }
    }
    __syncthreads();
  }

#pragma unroll
  for (int m = 0; m < M_; ++m)
#pragma unroll
    for (int c = 0; c < 4; ++c)
      red[(kq * M_ + m) * DK_ + d4 * 4 + c] = acc[m][c];
  __syncthreads();

  if (tid < DK_) {
    const int d = tid;
    for (int m = 0; m < M_; ++m) {
      float s = 0.f;
#pragma unroll
      for (int g = 0; g < 16; ++g) s += red[(g * M_ + m) * DK_ + d];
      part[((size_t)bh * SPLIT_ + sp) * SVROW_ + m * DK_ + d] = s;
    }
  }
}

// =====================================================================
// Kernel 2: fold the k-split partials and add the per-m bias.
// =====================================================================
__global__ void k_shrink_reduce(const float* __restrict__ part,
                                const float* __restrict__ bias,
                                float* __restrict__ sv) {
  const int idx = blockIdx.x * blockDim.x + threadIdx.x;   // [0, BH*512)
  if (idx >= BH_ * SVROW_) return;
  const int bh = idx >> 9;         // /512
  const int md = idx & 511;
  const int m = md >> 6;
  float s = bias[m];
#pragma unroll
  for (int sp = 0; sp < SPLIT_; ++sp)
    s += part[((size_t)bh * SPLIT_ + sp) * SVROW_ + md];
  sv[idx] = s;
}

// =====================================================================
// Kernel 3: cluster heads + loss (tiny; single block).
// =====================================================================
__global__ void k_cluster(const float* __restrict__ Ks,   // SV[t=0]
                          const float* __restrict__ Qs,   // SV[t=2]
                          const float* __restrict__ pkw, const float* __restrict__ pkb,
                          const float* __restrict__ pqw, const float* __restrict__ pqb,
                          const float* __restrict__ qpw, const float* __restrict__ qpb,
                          const float* __restrict__ kpw, const float* __restrict__ kpb,
                          float* __restrict__ out, int outN,
                          int* __restrict__ inds) {
  __shared__ float red[256];
  __shared__ float dots[24];
  const int tid = threadIdx.x;

  float acc[24];
#pragma unroll
  for (int i = 0; i < 24; ++i) acc[i] = 0.f;

  for (int e = tid; e < 8192; e += 256) {
    float wk0 = pkw[e], wk1 = pkw[8192 + e], wk2 = pkw[2 * 8192 + e];
    float wq0 = pqw[e], wq1 = pqw[8192 + e], wq2 = pqw[2 * 8192 + e];
#pragma unroll
    for (int b = 0; b < B_; ++b) {
      const float ks = Ks[(size_t)b * 8192 + e];
      const float qs = Qs[(size_t)b * 8192 + e];
      acc[b * 3 + 0]      += wk0 * ks; acc[b * 3 + 1]      += wk1 * ks; acc[b * 3 + 2]      += wk2 * ks;
      acc[12 + b * 3 + 0] += wq0 * qs; acc[12 + b * 3 + 1] += wq1 * qs; acc[12 + b * 3 + 2] += wq2 * qs;
    }
  }
  for (int di = 0; di < 24; ++di) {
    red[tid] = acc[di];
    __syncthreads();
    for (int s = 128; s > 0; s >>= 1) { if (tid < s) red[tid] += red[tid + s]; __syncthreads(); }
    if (tid == 0) dots[di] = red[0];
    __syncthreads();
  }

  if (tid == 0) {
    float ckp[B_][NC_], cqp[B_][NC_], cq[B_][NC_], ck[B_][NC_];
    for (int b = 0; b < B_; ++b)
      for (int c = 0; c < NC_; ++c) {
        ckp[b][c] = fmaxf(dots[b * 3 + c] + pkb[c], 0.f);
        cqp[b][c] = fmaxf(dots[12 + b * 3 + c] + pqb[c], 0.f);
      }
    for (int b = 0; b < B_; ++b) {
      float lq[NC_], lk[NC_], mq = -1e30f, mk = -1e30f;
      for (int c = 0; c < NC_; ++c) {
        float aq = qpb[c], ak = kpb[c];
        for (int j = 0; j < NC_; ++j) { aq += cqp[b][j] * qpw[c * 3 + j]; ak += ckp[b][j] * kpw[c * 3 + j]; }
        lq[c] = aq; lk[c] = ak; mq = fmaxf(mq, aq); mk = fmaxf(mk, ak);
      }
      float sq = 0.f, sk = 0.f;
      for (int c = 0; c < NC_; ++c) { cq[b][c] = expf(lq[c] - mq); ck[b][c] = expf(lk[c] - mk); sq += cq[b][c]; sk += ck[b][c]; }
      for (int c = 0; c < NC_; ++c) { cq[b][c] /= sq; ck[b][c] /= sk; }
    }
    float mu[NC_], sigma[NC_];
    for (int c = 0; c < NC_; ++c) {
      float s = 0.f; for (int b = 0; b < B_; ++b) s += cq[b][c];
      mu[c] = s * 0.25f;
      float mk = 0.f; for (int b = 0; b < B_; ++b) mk += ck[b][c];
      mk *= 0.25f;
      float var = 0.f; for (int b = 0; b < B_; ++b) { float d = ck[b][c] - mk; var += d * d; }
      var *= (1.f / 3.f);                       // ddof=1
      sigma[c] = logf(1.f + expf(sqrtf(var)));  // softplus(std)
    }
    const float half_log2pi = 0.5f * logf(2.f * 3.14159265358979323846f);
    float sumlogp = 0.f;
    for (int b = 0; b < B_; ++b)
      for (int c = 0; c < NC_; ++c) {
        float z = (ck[b][c] - mu[c]) / sigma[c];
        sumlogp += -0.5f * z * z - logf(sigma[c]) - half_log2pi;
      }
    float lossA = -sumlogp / 12.f;
    float ce = 0.f;
    for (int b = 0; b < B_; ++b) {
      float mx = fmaxf(fmaxf(cq[b][0], cq[b][1]), cq[b][2]);
      float lse = 0.f;
      for (int c = 0; c < NC_; ++c) lse += expf(cq[b][c] - mx);
      lse = mx + logf(lse);
      float s = 0.f;
      for (int c = 0; c < NC_; ++c) s += cq[b][c] * (cq[b][c] - lse);
      ce += -s;
    }
    ce *= 0.25f;
    out[outN - 1] = lossA + ce;

    for (int b = 0; b < B_; ++b) {
      int aq = 0, ak = 0;
      for (int c = 1; c < NC_; ++c) { if (cq[b][c] > cq[b][aq]) aq = c; if (ck[b][c] > ck[b][ak]) ak = c; }
      inds[b] = aq; inds[4 + b] = ak;
    }
  }
}

// =====================================================================
// Kernel 4: the scattered score row.  Last-write-wins scatter -> row j*=L-1.
// Compute batch-0 scores for the 16-row tile containing j* via f16 WMMA
// (Q[0,h,4080:4096,:64] x Ks[0,h]^T, K=64 as two K=32 WMMA steps).
// One wave per head; EXEC all-ones for the WMMA.
// =====================================================================
__global__ void k_scatrow(const float* __restrict__ Q,
                          const float* __restrict__ SVk,      // Ks: [BH][M][DK]
                          const int* __restrict__ inds,
                          float* __restrict__ scat) {
  __shared__ float smem[16 * 16];
  const int h = blockIdx.x;
  const int lane = threadIdx.x;
  const int row = 4080 + (lane & 15);
  const float* qrow = Q + ((size_t)h * L_ + row) * DK_;       // batch 0
  const int n = lane & 15;
  const bool nvalid = n < M_;
  const float* ksrow = SVk + (size_t)h * SVROW_ + n * DK_;    // Ks[0,h,n,:]

  v8f c = {};
#pragma unroll
  for (int chunk = 0; chunk < 2; ++chunk) {
    // A: 16x32 f16, lane (l%16)=M; lanes<16 hold K in {0..7,16..23}, else {8..15,24..31}
    v16h a;
    const int kbA = (lane < 16) ? 0 : 8;
#pragma unroll
    for (int v = 0; v < 4; ++v) {
      const int k = kbA + 2 * v;
      a[2 * v]     = (_Float16)qrow[chunk * 32 + k];
      a[2 * v + 1] = (_Float16)qrow[chunk * 32 + k + 1];
    }
#pragma unroll
    for (int v = 4; v < 8; ++v) {
      const int k = 16 + kbA + 2 * (v - 4);
      a[2 * v]     = (_Float16)qrow[chunk * 32 + k];
      a[2 * v + 1] = (_Float16)qrow[chunk * 32 + k + 1];
    }
    // B: 32x16 f16, lane (l%16)=N; lanes<16 hold K in {0..7,16..23}, else {8..15,24..31}
    v16h bv;
    const int kbB = (lane < 16) ? 0 : 8;
#pragma unroll
    for (int j = 0; j < 8; ++j) {
      const int k = (j < 4) ? (kbB + 2 * j) : (16 + kbB + 2 * (j - 4));
      bv[2 * j]     = nvalid ? (_Float16)ksrow[chunk * 32 + k]     : (_Float16)0;
      bv[2 * j + 1] = nvalid ? (_Float16)ksrow[chunk * 32 + k + 1] : (_Float16)0;
    }
    c = __builtin_amdgcn_wmma_f32_16x16x32_f16(false, a, false, bv, (short)0, c, false, false);
  }

  // dump C (VGPR j: lanes<16 -> M=j, lanes>=16 -> M=j+8; N = l%16) to LDS
#pragma unroll
  for (int j = 0; j < 8; ++j)
    smem[(j + ((lane < 16) ? 0 : 8)) * 16 + (lane & 15)] = c[j];
  __syncthreads();

  if (lane == 0) {
    const int iq0 = inds[0], ik0 = inds[4];
    const int qglob = L_ - 1;                       // last-write-wins scatter row
    for (int i = 0; i < NC_; ++i) {
      const float mq = (iq0 == i + 1) ? 0.f : 1.f;
      const float mk = (ik0 == i + 1) ? 0.f : 1.f;
      float s[M_], mx = -1e30f;
      for (int k = 0; k < M_; ++k) {
        float v = smem[15 * 16 + k] * 0.125f * mq * mk;   // scale = 1/sqrt(64)
        v = (k <= qglob) ? -1e9f : v;                     // tril mask row j*
        s[k] = v; mx = fmaxf(mx, v);
      }
      float den = 0.f;
      for (int k = 0; k < M_; ++k) { s[k] = expf(s[k] - mx); den += s[k]; }
      for (int k = 0; k < M_; ++k) scat[(i * H_ + h) * M_ + k] = s[k] / den;
    }
  }
}

// =====================================================================
// Kernel 5: attn + context.  Per (b,h): attn row q is
//   softmax(max_i scat[i,h,:])  if q == inds_q[b],  uniform 1/8 otherwise.
// context tile (16x64) = attn_tile(16x8) x Vs(8x64) via f16 WMMA.
// 8 waves/block, each wave owns one 16-row q-tile; store-bandwidth bound.
// =====================================================================
__global__ void k_context(const float* __restrict__ SVv,     // Vs: [BH][M][DK]
                          const float* __restrict__ scat,
                          const int* __restrict__ inds,
                          float* __restrict__ out) {
  __shared__ float vs[M_ * DK_];
  __shared__ float attn_sp[M_];
  __shared__ int rq_s;

  const int bh = blockIdx.x;
  const int b = bh >> 4, h = bh & 15;
  const int tid = threadIdx.x;

  for (int idx = tid; idx < M_ * DK_; idx += blockDim.x)
    vs[idx] = SVv[(size_t)bh * SVROW_ + idx];
  if (tid == 0) {
    rq_s = inds[b];
    float f[M_], mx = -1e30f;
    for (int k = 0; k < M_; ++k) {
      float v = scat[(0 * H_ + h) * M_ + k];
      v = fmaxf(v, scat[(1 * H_ + h) * M_ + k]);
      v = fmaxf(v, scat[(2 * H_ + h) * M_ + k]);
      f[k] = v; mx = fmaxf(mx, v);
    }
    float den = 0.f;
    for (int k = 0; k < M_; ++k) { f[k] = expf(f[k] - mx); den += f[k]; }
    for (int k = 0; k < M_; ++k) attn_sp[k] = f[k] / den;
  }
  __syncthreads();

  const int wave = tid >> 5;
  const int lane = tid & 31;
  const int qtile = blockIdx.y * 8 + wave;
  const int qbase = qtile * 16;
  const int r = qbase + (lane & 15);
  const int rq = rq_s;

  // per-lane attn row (A-matrix row M = lane%16)
  float arow[M_];
#pragma unroll
  for (int k = 0; k < M_; ++k) arow[k] = (r == rq) ? attn_sp[k] : 0.125f;

  // A: 16x32 f16, only K<8 nonzero.  lanes<16 hold K {0..7,16..23} -> {arow,0};
  // lanes>=16 hold K {8..15,24..31} -> all zero.
  v16h a;
#pragma unroll
  for (int i = 0; i < 16; ++i) a[i] = (_Float16)0;
  if (lane < 16) {
#pragma unroll
    for (int v = 0; v < 4; ++v) { a[2 * v] = (_Float16)arow[2 * v]; a[2 * v + 1] = (_Float16)arow[2 * v + 1]; }
  }

  const int col0 = lane & 15;
#pragma unroll
  for (int nt = 0; nt < 4; ++nt) {
    // B: 32x16 f16; lanes<16 hold K {0..7,16..23} (K<8 real), lanes>=16 hold K {8..15,24..31} -> zero
    v16h bv;
#pragma unroll
    for (int i = 0; i < 16; ++i) bv[i] = (_Float16)0;
    if (lane < 16) {
#pragma unroll
      for (int j = 0; j < 4; ++j) {           // K = 0..7
        bv[2 * j]     = (_Float16)vs[(2 * j) * DK_ + nt * 16 + col0];
        bv[2 * j + 1] = (_Float16)vs[(2 * j + 1) * DK_ + nt * 16 + col0];
      }
    }
    v8f c = {};
    c = __builtin_amdgcn_wmma_f32_16x16x32_f16(false, a, false, bv, (short)0, c, false, false);
#pragma unroll
    for (int j = 0; j < 8; ++j) {
      const int m = j + ((lane < 16) ? 0 : 8);
      out[((size_t)bh * L_ + qbase + m) * DK_ + nt * 16 + col0] = c[j];
    }
  }
}

// =====================================================================
extern "C" void kernel_launch(void* const* d_in, const int* in_sizes, int n_in,
                              void* d_out, int out_size, void* d_ws, size_t ws_size,
                              hipStream_t stream) {
  const float* Q   = (const float*)d_in[0];
  const float* K   = (const float*)d_in[1];
  const float* V   = (const float*)d_in[2];
  const float* skw = (const float*)d_in[3];
  const float* skb = (const float*)d_in[4];
  const float* svw = (const float*)d_in[5];
  const float* svb = (const float*)d_in[6];
  const float* sqw = (const float*)d_in[7];
  const float* sqb = (const float*)d_in[8];
  const float* pkw = (const float*)d_in[9];
  const float* pkb = (const float*)d_in[10];
  const float* pqw = (const float*)d_in[11];
  const float* pqb = (const float*)d_in[12];
  const float* qpw = (const float*)d_in[13];
  const float* qpb = (const float*)d_in[14];
  const float* kpw = (const float*)d_in[15];
  const float* kpb = (const float*)d_in[16];

  float* wsf = (float*)d_ws;
  float* part0 = wsf + 0 * (BH_ * SPLIT_ * SVROW_);
  float* part1 = wsf + 1 * (BH_ * SPLIT_ * SVROW_);
  float* part2 = wsf + 2 * (BH_ * SPLIT_ * SVROW_);
  float* sv0 = wsf + SV_OFF + 0 * (BH_ * SVROW_);   // Ks
  float* sv1 = wsf + SV_OFF + 1 * (BH_ * SVROW_);   // Vs
  float* sv2 = wsf + SV_OFF + 2 * (BH_ * SVROW_);   // Qs
  float* scat = wsf + SCAT_OFF;
  int* inds = (int*)(wsf + INDS_OFF);

  // 1) shrink K, V, Q  (192 MiB streamed once; HBM-bound)
  k_shrink_partial<<<dim3(BH_, SPLIT_), 256, 0, stream>>>(K, skw, part0);
  k_shrink_partial<<<dim3(BH_, SPLIT_), 256, 0, stream>>>(V, svw, part1);
  k_shrink_partial<<<dim3(BH_, SPLIT_), 256, 0, stream>>>(Q, sqw, part2);
  k_shrink_reduce<<<(BH_ * SVROW_ + 255) / 256, 256, 0, stream>>>(part0, skb, sv0);
  k_shrink_reduce<<<(BH_ * SVROW_ + 255) / 256, 256, 0, stream>>>(part1, svb, sv1);
  k_shrink_reduce<<<(BH_ * SVROW_ + 255) / 256, 256, 0, stream>>>(part2, sqb, sv2);

  // 2) cluster heads, loss, argmax indices
  k_cluster<<<1, 256, 0, stream>>>(sv0, sv2, pkw, pkb, pqw, pqb,
                                   qpw, qpb, kpw, kpb,
                                   (float*)d_out, out_size, inds);

  // 3) scattered score row (WMMA f16 score tile + masked softmax)
  k_scatrow<<<H_, 32, 0, stream>>>(Q, sv0, inds, scat);

  // 4) attn + context GEMM (WMMA f16, store-bound 64 MiB write)
  k_context<<<dim3(BH_, 32), 256, 0, stream>>>(sv1, scat, inds, (float*)d_out);
}
